// Attention2dDownsample_60730837565715
// MI455X (gfx1250) — matrix operand order repaired and checked by reading, not compiled, not run on hardware
//
#include <hip/hip_runtime.h>
#include <hip/hip_bf16.h>
#include <math.h>

// ---------------- problem constants ----------------
#define BB    32
#define DIM_  384
#define KEYD  32
#define NHEAD 8
#define NKEY  1024   // 32x32
#define NQ    256    // 16x16
#define KH_   256    // KEYD*NHEAD
#define DV_   128
#define DH_   1024   // DV*NHEAD
#define OUTD  768
#define SCALE_ 0.17677669529663687f  // 32^-0.5

// GEMM tiling
#define BK    64     // k-chunk staged per TDM transfer
#define BROWS 128    // B-panel rows (pixels) per block = 8 waves x 16 cols

// ---------------- WMMA plumbing (gfx1250, wave32) ----------------
typedef __attribute__((ext_vector_type(16))) __bf16 v16bf;
typedef __attribute__((ext_vector_type(8)))  __bf16 v8bf;
typedef __attribute__((ext_vector_type(8)))  float  v8f;
typedef __attribute__((ext_vector_type(4)))  unsigned int u32x4;
typedef __attribute__((ext_vector_type(8)))  unsigned int u32x8;

static __device__ __forceinline__ v8f wmma_bf16(v16bf a, v16bf b, v8f c) {
  return __builtin_amdgcn_wmma_f32_16x16x32_bf16(false, a, false, b, (short)0, c,
                                                 false, false);
}

// A fragment (row-major 16x32 tile): row = lane&15,
// halves 0-7 -> K = kb..kb+7, halves 8-15 -> K = kb+16..kb+23, kb = lane<16?0:8
static __device__ __forceinline__ v16bf load_frag_a(const __bf16* base, int lda, int lane) {
  const __bf16* p = base + (lane & 15) * lda + ((lane < 16) ? 0 : 8);
  v16bf a;
  ((v8bf*)&a)[0] = *(const v8bf*)(p);
  ((v8bf*)&a)[1] = *(const v8bf*)(p + 16);
  return a;
}
// B fragment (B col-major, stored as Bt[n][k] row-major, 32x16 tile):
// col = lane&15, halves 0-7 -> K = kb..kb+7, halves 8-15 -> K = kb+16..kb+23, kb = lane<16?0:8
static __device__ __forceinline__ v16bf load_frag_b(const __bf16* base, int ldb, int lane) {
  const __bf16* p = base + (lane & 15) * ldb + ((lane < 16) ? 0 : 8);
  v16bf b;
  ((v8bf*)&b)[0] = *(const v8bf*)(p);
  ((v8bf*)&b)[1] = *(const v8bf*)(p + 16);
  return b;
}

// ---------------- Tensor Data Mover: 2-D tile global -> LDS ----------------
// D# per CDNA5 ISA. 2-byte elements.
// tile = rowElems x rows, row stride = strideElems, packed contiguously in LDS.
static __device__ __forceinline__ void tdm_load_2d(const void* gptr, unsigned ldsByteOff,
                                                   unsigned rowElems, unsigned rows,
                                                   unsigned strideElems) {
  unsigned long long ga = (unsigned long long)(size_t)gptr;
  u32x4 g0;
  g0[0] = 1u;                                               // count=1, user desc
  g0[1] = ldsByteOff;                                       // lds_addr
  g0[2] = (unsigned)ga;                                     // global_addr[31:0]
  g0[3] = ((unsigned)(ga >> 32) & 0x01FFFFFFu) | (2u << 30); // addr[56:32] | type=2
  u32x8 g1;
  g1[0] = 1u << 16;                                         // data_size=1 (2 bytes)
  g1[1] = (strideElems & 0xFFFFu) << 16;                    // tensor_dim0[15:0]
  g1[2] = (strideElems >> 16) | (rows << 16);               // dim0 hi | tensor_dim1 lo
  g1[3] = rowElems << 16;                                   // dim1 hi=0 | tile_dim0
  g1[4] = rows;                                             // tile_dim1 | tile_dim2=0
  g1[5] = strideElems;                                      // tensor_dim0_stride[31:0]
  g1[6] = 0;                                                // stride hi | dim1_stride lo
  g1[7] = 0;
  asm volatile("tensor_load_to_lds %0, %1" :: "s"(g0), "s"(g1) : "memory");
}

// ---------------- prep kernels ----------------
__global__ void cvt_f32_bf16(const float* __restrict__ s, __bf16* __restrict__ d, int n) {
  int i = blockIdx.x * blockDim.x + threadIdx.x;
  if (i < n) d[i] = (__bf16)s[i];
}

// xt[b,p,c] = x[b,c,p]  (pixel-major, channel contiguous -> WMMA B operand)
__global__ void prep_xt(const float* __restrict__ x, __bf16* __restrict__ xt) {
  int p = blockIdx.x & (NKEY - 1);
  int b = blockIdx.x >> 10;
  int c = threadIdx.x;  // blockDim = 384
  xt[((long)b * NKEY + p) * DIM_ + c] = (__bf16)x[((long)b * DIM_ + c) * NKEY + p];
}

// qin[b,p2,c] = dwconv3x3_s2(x)[b,c,p2] + local_b[c] + x[b,c,2i,2j]
__global__ void prep_qin(const float* __restrict__ x, const float* __restrict__ lw,
                         const float* __restrict__ lb, __bf16* __restrict__ qin) {
  int p2 = blockIdx.x & (NQ - 1);
  int b  = blockIdx.x >> 8;
  int c  = threadIdx.x;  // blockDim = 384
  int i = p2 >> 4, j = p2 & 15;
  const float* xb = x + ((long)b * DIM_ + c) * NKEY;
  float acc = lb[c];
#pragma unroll
  for (int dy = -1; dy <= 1; ++dy)
#pragma unroll
    for (int dx = -1; dx <= 1; ++dx) {
      int yy = 2 * i + dy, xx = 2 * j + dx;
      if (yy >= 0 && yy < 32 && xx >= 0 && xx < 32)
        acc += xb[yy * 32 + xx] * lw[c * 9 + (dy + 1) * 3 + (dx + 1)];
    }
  acc += xb[(2 * i) * 32 + 2 * j];  // pool_q (stride-2 sample)
  qin[((long)b * NQ + p2) * DIM_ + c] = (__bf16)acc;
}

// ---------------- TDM-staged pointwise GEMM ----------------
// Y[o,p] = s[o]*sum_c W[o,c] X[c,p] + b[o].
// One block (8 waves) = 16 output rows x 128 output cols.
// Wave 0 DMAs A-strip (16xBK) and B-panel (128xBK) per k-chunk, double-buffered.
template <typename OutT>
__global__ __launch_bounds__(256) void gemm_bf16_tdm(
    const __bf16* __restrict__ A, const __bf16* __restrict__ Bt,
    const float* __restrict__ sc, const float* __restrict__ bi,
    OutT* __restrict__ out, int M, int K, int Npix,
    long bStrideB, long bStrideOut, int strideO, int strideP) {
  __shared__ __align__(16) __bf16 smem[2 * BROWS * BK + 2 * 16 * BK];  // 36 KB
  const unsigned AOFF = 2 * BROWS * BK;  // element offset of A buffers

  int tilesM  = M >> 4;
  int nBlocks = Npix >> 7;
  int blk = blockIdx.x;
  int b   = blk / (tilesM * nBlocks);
  int t   = blk - b * (tilesM * nBlocks);
  int mt  = t / nBlocks;
  int nb  = t - mt * nBlocks;
  int tid = threadIdx.x, wave = tid >> 5, lane = tid & 31;

  const __bf16* Ab = A + (long)(mt * 16) * K;
  const __bf16* Bb = Bt + (long)b * bStrideB + (long)(nb * BROWS) * K;
  int nC = K / BK;

  if (wave == 0) {  // prologue: chunk 0 (TDM ignores EXEC; one issue per wave)
    tdm_load_2d(Bb, 0, BK, BROWS, (unsigned)K);
    tdm_load_2d(Ab, AOFF * 2, BK, 16, (unsigned)K);
  }

  v8f acc = {0.f, 0.f, 0.f, 0.f, 0.f, 0.f, 0.f, 0.f};
  for (int c = 0; c < nC; ++c) {
    int par = c & 1;
    if (wave == 0) {
      if (c + 1 < nC) {
        int np = (c + 1) & 1;
        tdm_load_2d(Bb + (c + 1) * BK, (unsigned)(np * BROWS * BK) * 2, BK, BROWS, (unsigned)K);
        tdm_load_2d(Ab + (c + 1) * BK, (AOFF + np * 16 * BK) * 2, BK, 16, (unsigned)K);
        __builtin_amdgcn_s_wait_tensorcnt(2);  // in-order: chunk c has landed
      } else {
        __builtin_amdgcn_s_wait_tensorcnt(0);
      }
    }
    __syncthreads();
    const __bf16* Bp = smem + par * BROWS * BK + wave * 16 * BK;
    const __bf16* Ap = smem + AOFF + par * 16 * BK;
    acc = wmma_bf16(load_frag_a(Ap, BK, lane),      load_frag_b(Bp, BK, lane),      acc);
    acc = wmma_bf16(load_frag_a(Ap + 32, BK, lane), load_frag_b(Bp + 32, BK, lane), acc);
    __syncthreads();  // protect buffer 'par' before it is overwritten next round
  }

  int mrow = mt * 16 + ((lane < 16) ? 0 : 8);
  int ncol = nb * BROWS + wave * 16 + (lane & 15);
  OutT* ob = out + (long)b * bStrideOut;
#pragma unroll
  for (int r = 0; r < 8; ++r) {
    int m = mrow + r;
    float y = acc[r] * sc[m] + bi[m];
    ob[(long)m * strideO + (long)ncol * strideP] = (OutT)y;
  }
}

// ---------------- attention: one block per (b, head, 16-row q tile) ----------------
__global__ __launch_bounds__(256) void attn_kernel(
    const __bf16* __restrict__ qt,   // [b, p2, KH]   (head dim contiguous)
    const __bf16* __restrict__ kt,   // [b, p,  KH]
    const __bf16* __restrict__ vb,   // [b, ch, p]    (pixel contiguous)
    const float* __restrict__ abias, // [H, NKEY]
    const int* __restrict__ bidx,    // [NQ, NKEY]
    float* __restrict__ obuf) {      // [b, ch, p2]
  __shared__ __align__(16) float  sA[16 * NKEY];   // logits / exp, f32 (64KB)
  __shared__ __align__(16) __bf16 sP[16 * NKEY];   // probs bf16 (32KB)
  __shared__ float sred[16][17];
  int blk = blockIdx.x;
  int mt = blk & 15;
  int h  = (blk >> 4) & 7;
  int b  = blk >> 7;
  int tid = threadIdx.x, wave = tid >> 5, lane = tid & 31;

  // ---- QK^T * scale + rel-pos bias -> sA ----
  const __bf16* qbase = qt + (long)b * NQ * KH_ + (long)(mt * 16) * KH_ + h * KEYD;
  const __bf16* kbase = kt + (long)b * NKEY * KH_ + h * KEYD;
  v16bf afrag = load_frag_a(qbase, KH_, lane);  // K = 32 -> single WMMA step
  for (int nt = wave; nt < NKEY / 16; nt += 8) {
    v16bf bfrag = load_frag_b(kbase + (long)(nt * 16) * KH_, KH_, lane);
    v8f c = {0.f, 0.f, 0.f, 0.f, 0.f, 0.f, 0.f, 0.f};
    c = wmma_bf16(afrag, bfrag, c);
    int mrow = (lane < 16) ? 0 : 8;
    int n = nt * 16 + (lane & 15);
#pragma unroll
    for (int r = 0; r < 8; ++r) {
      int m = mrow + r;                 // row within tile
      int mg = mt * 16 + m;             // global q index
      int idx = bidx[mg * NKEY + n];
      sA[m * NKEY + n] = c[r] * SCALE_ + abias[h * NKEY + idx];
    }
  }
  __syncthreads();

  // ---- softmax over keys: 16 threads per row ----
  int r = tid >> 4, l16 = tid & 15;
  float mx = -1e30f;
  for (int j = l16; j < NKEY; j += 16) mx = fmaxf(mx, sA[r * NKEY + j]);
  sred[r][l16] = mx;
  __syncthreads();
  if (l16 == 0) {
    float m2 = sred[r][0];
    for (int t = 1; t < 16; ++t) m2 = fmaxf(m2, sred[r][t]);
    sred[r][16] = m2;
  }
  __syncthreads();
  float rowmax = sred[r][16];
  float sm = 0.f;
  for (int j = l16; j < NKEY; j += 16) {
    float e = expf(sA[r * NKEY + j] - rowmax);
    sA[r * NKEY + j] = e;
    sm += e;
  }
  __syncthreads();
  sred[r][l16] = sm;
  __syncthreads();
  if (l16 == 0) {
    float s2 = 0.f;
    for (int t = 0; t < 16; ++t) s2 += sred[r][t];
    sred[r][16] = s2;
  }
  __syncthreads();
  float inv = 1.f / sred[r][16];
  for (int j = l16; j < NKEY; j += 16)
    sP[r * NKEY + j] = (__bf16)(sA[r * NKEY + j] * inv);
  __syncthreads();

  // ---- attn @ V : (16 x 1024) x (1024 x 16dv), one dv tile per wave ----
  v8f acc = {0.f, 0.f, 0.f, 0.f, 0.f, 0.f, 0.f, 0.f};
  const __bf16* vbase = vb + (long)b * DH_ * NKEY + (long)(h * DV_ + wave * 16) * NKEY;
  for (int k = 0; k < NKEY; k += 32) {
    __builtin_prefetch(vbase + k + 256, 0, 1);  // global_prefetch_b8 on streamed V
    v16bf a2 = load_frag_a((const __bf16*)sP + k, NKEY, lane);  // ds_load_b128 from LDS
    v16bf b2 = load_frag_b(vbase + k, NKEY, lane);
    acc = wmma_bf16(a2, b2, acc);
  }
  int mrow = (lane < 16) ? 0 : 8;
  int ch = h * DV_ + wave * 16 + (lane & 15);
  float* ob = obuf + (long)b * DH_ * NQ + (long)ch * NQ + mt * 16 + mrow;
#pragma unroll
  for (int rr = 0; rr < 8; ++rr) ob[rr] = acc[rr];
}

// ---------------- v_local dwconv + residual add + exact GELU ----------------
__global__ void vlocal_gelu(const __bf16* __restrict__ vb, const float* __restrict__ vw,
                            const float* __restrict__ vs, const float* __restrict__ vbi,
                            const float* __restrict__ obuf, __bf16* __restrict__ oact) {
  int p2 = blockIdx.x & (NQ - 1);
  int b  = blockIdx.x >> 8;
  int i = p2 >> 4, j = p2 & 15;
  for (int ch = threadIdx.x; ch < DH_; ch += blockDim.x) {
    const __bf16* vbase = vb + ((long)b * DH_ + ch) * NKEY;
    float acc = 0.f;
#pragma unroll
    for (int dy = -1; dy <= 1; ++dy)
#pragma unroll
      for (int dx = -1; dx <= 1; ++dx) {
        int yy = 2 * i + dy, xx = 2 * j + dx;
        if (yy >= 0 && yy < 32 && xx >= 0 && xx < 32)
          acc += (float)vbase[yy * 32 + xx] * vw[ch * 9 + (dy + 1) * 3 + (dx + 1)];
      }
    acc = acc * vs[ch] + vbi[ch];
    float o = obuf[((long)b * DH_ + ch) * NQ + p2] + acc;
    float g = 0.5f * o * (1.f + erff(o * 0.70710678118654752f));
    oact[((long)b * NQ + p2) * DH_ + ch] = (__bf16)g;  // pixel-major for final GEMM B
  }
}

// ---------------- host launch ----------------
extern "C" void kernel_launch(void* const* d_in, const int* in_sizes, int n_in,
                              void* d_out, int out_size, void* d_ws, size_t ws_size,
                              hipStream_t stream) {
  const float* x       = (const float*)d_in[0];
  const float* local_w = (const float*)d_in[1];
  const float* local_b = (const float*)d_in[2];
  const float* qproj_w = (const float*)d_in[3];
  const float* qproj_s = (const float*)d_in[4];
  const float* qproj_b = (const float*)d_in[5];
  const float* k_w     = (const float*)d_in[6];
  const float* k_s     = (const float*)d_in[7];
  const float* k_b     = (const float*)d_in[8];
  const float* v_w     = (const float*)d_in[9];
  const float* v_s     = (const float*)d_in[10];
  const float* v_b     = (const float*)d_in[11];
  const float* vloc_w  = (const float*)d_in[12];
  const float* vloc_s  = (const float*)d_in[13];
  const float* vloc_b  = (const float*)d_in[14];
  const float* proj_w  = (const float*)d_in[15];
  const float* proj_s  = (const float*)d_in[16];
  const float* proj_b  = (const float*)d_in[17];
  const float* abias   = (const float*)d_in[18];
  const int*   bidx    = (const int*)d_in[19];
  float* out = (float*)d_out;

  char* ws = (char*)d_ws;
  size_t off = 0;
  auto carve = [&](size_t bytes) { char* p = ws + off; off += (bytes + 255) & ~(size_t)255; return p; };
  __bf16* XT  = (__bf16*)carve((size_t)BB * NKEY * DIM_ * 2);  // x transposed
  __bf16* QIN = (__bf16*)carve((size_t)BB * NQ * DIM_ * 2);    // q input
  __bf16* WK  = (__bf16*)carve((size_t)KH_ * DIM_ * 2);
  __bf16* WV  = (__bf16*)carve((size_t)DH_ * DIM_ * 2);
  __bf16* WQ  = (__bf16*)carve((size_t)KH_ * DIM_ * 2);
  __bf16* WP  = (__bf16*)carve((size_t)OUTD * DH_ * 2);
  __bf16* KT  = (__bf16*)carve((size_t)BB * NKEY * KH_ * 2);   // k, [b,p,o]
  __bf16* VB  = (__bf16*)carve((size_t)BB * DH_ * NKEY * 2);   // v, [b,o,p]
  __bf16* QT  = (__bf16*)carve((size_t)BB * NQ * KH_ * 2);     // q, [b,p2,o]
  float*  OB  = (float*)carve((size_t)BB * DH_ * NQ * 4);      // attn out, [b,o,p2]
  __bf16* OA  = (__bf16*)carve((size_t)BB * NQ * DH_ * 2);     // gelu out, [b,p2,o]

  // weights -> bf16
  cvt_f32_bf16<<<(KH_ * DIM_ + 255) / 256, 256, 0, stream>>>(k_w, WK, KH_ * DIM_);
  cvt_f32_bf16<<<(DH_ * DIM_ + 255) / 256, 256, 0, stream>>>(v_w, WV, DH_ * DIM_);
  cvt_f32_bf16<<<(KH_ * DIM_ + 255) / 256, 256, 0, stream>>>(qproj_w, WQ, KH_ * DIM_);
  cvt_f32_bf16<<<(OUTD * DH_ + 255) / 256, 256, 0, stream>>>(proj_w, WP, OUTD * DH_);

  prep_xt<<<BB * NKEY, DIM_, 0, stream>>>(x, XT);
  prep_qin<<<BB * NQ, DIM_, 0, stream>>>(x, local_w, local_b, QIN);

  // k projection: 256x384 @ 384x1024 per batch -> KT[b,p,o]
  gemm_bf16_tdm<__bf16><<<BB * (KH_ / 16) * (NKEY / 128), 256, 0, stream>>>(
      WK, XT, k_s, k_b, KT, KH_, DIM_, NKEY,
      (long)NKEY * DIM_, (long)NKEY * KH_, /*strideO=*/1, /*strideP=*/KH_);
  // v projection: 1024x384 @ 384x1024 -> VB[b,o,p]
  gemm_bf16_tdm<__bf16><<<BB * (DH_ / 16) * (NKEY / 128), 256, 0, stream>>>(
      WV, XT, v_s, v_b, VB, DH_, DIM_, NKEY,
      (long)NKEY * DIM_, (long)DH_ * NKEY, /*strideO=*/NKEY, /*strideP=*/1);
  // q projection: 256x384 @ 384x256 -> QT[b,p2,o]
  gemm_bf16_tdm<__bf16><<<BB * (KH_ / 16) * (NQ / 128), 256, 0, stream>>>(
      WQ, QIN, qproj_s, qproj_b, QT, KH_, DIM_, NQ,
      (long)NQ * DIM_, (long)NQ * KH_, /*strideO=*/1, /*strideP=*/KH_);

  attn_kernel<<<BB * NHEAD * (NQ / 16), 256, 0, stream>>>(QT, KT, VB, abias, bidx, OB);

  vlocal_gelu<<<BB * NQ, 256, 0, stream>>>(VB, vloc_w, vloc_s, vloc_b, OB, OA);

  // output projection: 768x1024 @ 1024x256 -> d_out[b,o,p2] f32
  gemm_bf16_tdm<float><<<BB * (OUTD / 16) * (NQ / 128), 256, 0, stream>>>(
      WP, OA, proj_s, proj_b, out, OUTD, DH_, NQ,
      (long)NQ * DH_, (long)OUTD * NQ, /*strideO=*/NQ, /*strideP=*/1);
}
